// NeuralODELayer_13967233647588
// MI455X (gfx1250) — compile-verified, matrix operand order checked by reading, not run on hardware
//
#include <hip/hip_runtime.h>
#include <hip/hip_bf16.h>

// ---------------------------------------------------------------------------
// Problem constants
// ---------------------------------------------------------------------------
#define BATCH   16384
#define IN_DIM  256
#define HID     512
#define MROWS   64          // rows per workgroup
#define LDA_S   (HID + 16)  // padded LDS stride (bf16 elems) for 512-wide bufs
#define LDA_P   (IN_DIM + 16)

typedef __attribute__((ext_vector_type(16))) __bf16 v16bf;
typedef __attribute__((ext_vector_type(8)))  __bf16 v8bf;
typedef __attribute__((ext_vector_type(4)))  __bf16 v4bf;
typedef __attribute__((ext_vector_type(8)))  float  v8f;

// ---------------------------------------------------------------------------
// Core WMMA tile loop: acc[4][4] += A(64xK, LDS bf16) @ Wt(N rows x K, bf16)^T
// Wt is stored transposed: Wt[n*K + k], so a "B" fragment for output column n
// is a K-contiguous load, identical addressing pattern to the A fragment.
//
// 16-bit A-matrix layout (ISA 7.12.2): lane = m (0..15) | m (16..31 = hi-K),
//   kbase = (lane>>4)*8, vec elems 0..7 <-> K kbase..kbase+7,
//   elems 8..15 <-> K 16+kbase .. 16+kbase+7. B mirrors with lane = n.
// ---------------------------------------------------------------------------
template<int KDIM, int LDA>
__device__ __forceinline__ void gemm_tiles(const __bf16* __restrict__ a_lds,
                                           const __bf16* __restrict__ Wt,
                                           v8f acc[4][4], int lane, int nbase)
{
    const int mlane = lane & 15;
    const int kbase = (lane >> 4) << 3;
#pragma unroll 1
    for (int kt = 0; kt < KDIM; kt += 32) {
        v16bf a[4];
#pragma unroll
        for (int mt = 0; mt < 4; ++mt) {
            const __bf16* p = a_lds + (mt * 16 + mlane) * LDA + kt + kbase;
            v8bf lo = *(const v8bf*)p;
            v8bf hi = *(const v8bf*)(p + 16);
            a[mt] = __builtin_shufflevector(lo, hi,
                    0,1,2,3,4,5,6,7,8,9,10,11,12,13,14,15);
        }
#pragma unroll
        for (int nt = 0; nt < 4; ++nt) {
            const __bf16* q = Wt + (size_t)(nbase + nt * 16 + mlane) * KDIM
                                 + kt + kbase;
            v8bf lo = *(const v8bf*)q;
            v8bf hi = *(const v8bf*)(q + 16);
            v16bf b = __builtin_shufflevector(lo, hi,
                    0,1,2,3,4,5,6,7,8,9,10,11,12,13,14,15);
#pragma unroll
            for (int mt = 0; mt < 4; ++mt) {
                acc[mt][nt] = __builtin_amdgcn_wmma_f32_16x16x32_bf16(
                    false, a[mt], false, b, (short)0, acc[mt][nt],
                    false, false);
            }
        }
    }
}

__device__ __forceinline__ void zero_acc(v8f acc[4][4])
{
    const v8f z = {0.f, 0.f, 0.f, 0.f, 0.f, 0.f, 0.f, 0.f};
#pragma unroll
    for (int i = 0; i < 4; ++i)
#pragma unroll
        for (int j = 0; j < 4; ++j)
            acc[i][j] = z;
}

// GEMM + bias + tanh, result (bf16) into an LDS activation buffer.
__device__ __forceinline__ void gemm_bias_tanh_to_lds(
    const __bf16* __restrict__ a_lds, const __bf16* __restrict__ Wt,
    const float* __restrict__ bias, __bf16* __restrict__ out_lds,
    int lane, int nbase, int mhi, int mlane)
{
    v8f acc[4][4];
    zero_acc(acc);
    gemm_tiles<HID, LDA_S>(a_lds, Wt, acc, lane, nbase);
#pragma unroll
    for (int mt = 0; mt < 4; ++mt)
#pragma unroll
        for (int nt = 0; nt < 4; ++nt)
#pragma unroll
            for (int r = 0; r < 8; ++r) {
                const int col = nbase + nt * 16 + mlane;
                const float v = acc[mt][nt][r] + bias[col];
                out_lds[(mt * 16 + mhi + r) * LDA_S + col] = (__bf16)tanhf(v);
            }
}

// ---------------------------------------------------------------------------
// Fused dopri5 stage kernel.
//   phase0: A_in = s + c1*k_a + ... (bf16, into LDS)
//   phase1: h1 = tanh(A_in @ Wt1 + b1)   (LDS -> LDS)
//   phase2: h2 = tanh(h1  @ Wt2 + b2)    (LDS -> LDS)
//   phase3: out = h2 @ Wt3 + b3
//     MODE 0: kout[g] = bf16(out)
//     MODE 1: sout[g] = s[g] + u1*k1 + u3*k3 + u4*k4 + u5*k5 + u6*out
// ---------------------------------------------------------------------------
template<int NK, int MODE>
__global__ __launch_bounds__(256) void ode_stage_kernel(
    const float* s,
    const __bf16* __restrict__ ka, const __bf16* __restrict__ kb,
    const __bf16* __restrict__ kc, const __bf16* __restrict__ kd,
    const __bf16* __restrict__ ke,
    float c1, float c2, float c3, float c4, float c5,
    const __bf16* __restrict__ Wt1, const float* __restrict__ bb1,
    const __bf16* __restrict__ Wt2, const float* __restrict__ bb2,
    const __bf16* __restrict__ Wt3, const float* __restrict__ bb3,
    __bf16* __restrict__ kout, float* sout,
    float u1, float u3, float u4, float u5, float u6)
{
    extern __shared__ __align__(16) char smem[];
    __bf16* lds_a = (__bf16*)smem;
    __bf16* lds_b = lds_a + MROWS * LDA_S;

    const int tid  = threadIdx.x;
    const int row0 = blockIdx.x << 6;

    // ---- phase 0: build stage input in LDS (bf16) ----
    for (int i = tid; i < MROWS * (HID / 4); i += 256) {
        const int r = i >> 7;
        const int c = (i & 127) << 2;
        const int g = (row0 + r) * HID + c;
        const float4 sv = *(const float4*)(s + g);
        float v0 = sv.x, v1 = sv.y, v2 = sv.z, v3 = sv.w;
        if constexpr (NK >= 1) { v4bf k = *(const v4bf*)(ka + g);
            v0 += c1*(float)k[0]; v1 += c1*(float)k[1];
            v2 += c1*(float)k[2]; v3 += c1*(float)k[3]; }
        if constexpr (NK >= 2) { v4bf k = *(const v4bf*)(kb + g);
            v0 += c2*(float)k[0]; v1 += c2*(float)k[1];
            v2 += c2*(float)k[2]; v3 += c2*(float)k[3]; }
        if constexpr (NK >= 3) { v4bf k = *(const v4bf*)(kc + g);
            v0 += c3*(float)k[0]; v1 += c3*(float)k[1];
            v2 += c3*(float)k[2]; v3 += c3*(float)k[3]; }
        if constexpr (NK >= 4) { v4bf k = *(const v4bf*)(kd + g);
            v0 += c4*(float)k[0]; v1 += c4*(float)k[1];
            v2 += c4*(float)k[2]; v3 += c4*(float)k[3]; }
        if constexpr (NK >= 5) { v4bf k = *(const v4bf*)(ke + g);
            v0 += c5*(float)k[0]; v1 += c5*(float)k[1];
            v2 += c5*(float)k[2]; v3 += c5*(float)k[3]; }
        v4bf o;
        o[0] = (__bf16)v0; o[1] = (__bf16)v1;
        o[2] = (__bf16)v2; o[3] = (__bf16)v3;
        *(v4bf*)(lds_a + r * LDA_S + c) = o;
    }
    __syncthreads();

    const int lane  = tid & 31;
    const int wave  = tid >> 5;
    const int nbase = wave << 6;
    const int mlane = lane & 15;
    const int mhi   = (lane >> 4) << 3;

    // ---- phase 1: lds_a -> lds_b ----
    gemm_bias_tanh_to_lds(lds_a, Wt1, bb1, lds_b, lane, nbase, mhi, mlane);
    __syncthreads();
    // ---- phase 2: lds_b -> lds_a ----
    gemm_bias_tanh_to_lds(lds_b, Wt2, bb2, lds_a, lane, nbase, mhi, mlane);
    __syncthreads();

    // ---- phase 3: lds_a @ Wt3 + b3 -> output ----
    {
        v8f acc[4][4];
        zero_acc(acc);
        gemm_tiles<HID, LDA_S>(lds_a, Wt3, acc, lane, nbase);
#pragma unroll
        for (int mt = 0; mt < 4; ++mt)
#pragma unroll
            for (int nt = 0; nt < 4; ++nt)
#pragma unroll
                for (int r = 0; r < 8; ++r) {
                    const int col = nbase + nt * 16 + mlane;
                    const int g   = (row0 + mt * 16 + mhi + r) * HID + col;
                    const float v = acc[mt][nt][r] + bb3[col];
                    if constexpr (MODE == 0) {
                        kout[g] = (__bf16)v;
                    } else {
                        float upd = s[g]
                                  + u1 * (float)ka[g] + u3 * (float)kc[g]
                                  + u4 * (float)kd[g] + u5 * (float)ke[g]
                                  + u6 * v;
                        sout[g] = upd;
                    }
                }
    }
}

// ---------------------------------------------------------------------------
// Input projection: s = y + u_t @ Wpt^T + bp   (K = 256)
// ---------------------------------------------------------------------------
__global__ __launch_bounds__(256) void proj_kernel(
    const float* __restrict__ y, const float* __restrict__ u_t,
    const __bf16* __restrict__ Wpt, const float* __restrict__ bp,
    float* __restrict__ s)
{
    extern __shared__ __align__(16) char smem[];
    __bf16* lds_a = (__bf16*)smem;

    const int tid  = threadIdx.x;
    const int row0 = blockIdx.x << 6;

    for (int i = tid; i < MROWS * (IN_DIM / 4); i += 256) {
        const int r = i >> 6;
        const int c = (i & 63) << 2;
        const float4 v = *(const float4*)(u_t + (row0 + r) * IN_DIM + c);
        v4bf o;
        o[0] = (__bf16)v.x; o[1] = (__bf16)v.y;
        o[2] = (__bf16)v.z; o[3] = (__bf16)v.w;
        *(v4bf*)(lds_a + r * LDA_P + c) = o;
    }
    __syncthreads();

    const int lane  = tid & 31;
    const int wave  = tid >> 5;
    const int nbase = wave << 6;
    const int mlane = lane & 15;
    const int mhi   = (lane >> 4) << 3;

    v8f acc[4][4];
    zero_acc(acc);
    gemm_tiles<IN_DIM, LDA_P>(lds_a, Wpt, acc, lane, nbase);
#pragma unroll
    for (int mt = 0; mt < 4; ++mt)
#pragma unroll
        for (int nt = 0; nt < 4; ++nt)
#pragma unroll
            for (int r = 0; r < 8; ++r) {
                const int col = nbase + nt * 16 + mlane;
                const int g   = (row0 + mt * 16 + mhi + r) * HID + col;
                s[g] = y[g] + acc[mt][nt][r] + bp[col];
            }
}

// ---------------------------------------------------------------------------
// Convert fp32 weights to bf16, transposed (Wt[n*K + k] = W[k*N + n]).
// ---------------------------------------------------------------------------
__global__ __launch_bounds__(256) void convert_weights_kernel(
    const float* __restrict__ Wp, const float* __restrict__ W1,
    const float* __restrict__ W2, const float* __restrict__ W3,
    __bf16* __restrict__ Wpt, __bf16* __restrict__ Wt1,
    __bf16* __restrict__ Wt2, __bf16* __restrict__ Wt3)
{
    const int idx = blockIdx.x * 256 + threadIdx.x;   // 512*512 threads
    const int n = idx >> 9;
    const int k = idx & 511;
    Wt1[n * HID + k] = (__bf16)W1[k * HID + n];
    Wt2[n * HID + k] = (__bf16)W2[k * HID + n];
    Wt3[n * HID + k] = (__bf16)W3[k * HID + n];
    if (k < IN_DIM)
        Wpt[n * IN_DIM + k] = (__bf16)Wp[k * HID + n];
}

// ---------------------------------------------------------------------------
// Host-side orchestration
// ---------------------------------------------------------------------------
extern "C" void kernel_launch(void* const* d_in, const int* in_sizes, int n_in,
                              void* d_out, int out_size, void* d_ws,
                              size_t ws_size, hipStream_t stream)
{
    const float* y   = (const float*)d_in[0];
    const float* u_t = (const float*)d_in[1];
    const float* Wp  = (const float*)d_in[2];
    const float* bp  = (const float*)d_in[3];
    const float* W1  = (const float*)d_in[4];
    const float* b1  = (const float*)d_in[5];
    const float* W2  = (const float*)d_in[6];
    const float* b2  = (const float*)d_in[7];
    const float* W3  = (const float*)d_in[8];
    const float* b3  = (const float*)d_in[9];
    float* out = (float*)d_out;

    constexpr size_t SB = (size_t)BATCH * HID * sizeof(float);   // state
    constexpr size_t KB = (size_t)BATCH * HID * sizeof(__bf16);  // one k buf

    char* ws = (char*)d_ws;
    float*  s  = (float*)ws;
    __bf16* k1 = (__bf16*)(ws + SB + 0 * KB);
    __bf16* k2 = (__bf16*)(ws + SB + 1 * KB);
    __bf16* k3 = (__bf16*)(ws + SB + 2 * KB);
    __bf16* k4 = (__bf16*)(ws + SB + 3 * KB);
    __bf16* k5 = (__bf16*)(ws + SB + 4 * KB);
    __bf16* Wpt = (__bf16*)(ws + SB + 5 * KB);
    __bf16* Wt1 = Wpt + (size_t)HID * IN_DIM;
    __bf16* Wt2 = Wt1 + (size_t)HID * HID;
    __bf16* Wt3 = Wt2 + (size_t)HID * HID;

    const int    nblk       = BATCH / MROWS;                   // 256
    const size_t stage_smem = (size_t)2 * MROWS * LDA_S * sizeof(__bf16);
    const size_t proj_smem  = (size_t)MROWS * LDA_P * sizeof(__bf16);

    convert_weights_kernel<<<(HID * HID) / 256, 256, 0, stream>>>(
        Wp, W1, W2, W3, Wpt, Wt1, Wt2, Wt3);
    proj_kernel<<<nblk, 256, proj_smem, stream>>>(y, u_t, Wpt, bp, s);

    const float h = 0.0125f;  // DT * N_STEPS_ARG / N_SOLVER_STEPS
    const float a21 = h * 0.2f;
    const float a31 = h * (3.f/40.f),  a32 = h * (9.f/40.f);
    const float a41 = h * (44.f/45.f), a42 = h * (-56.f/15.f),
                a43 = h * (32.f/9.f);
    const float a51 = h * (19372.f/6561.f), a52 = h * (-25360.f/2187.f),
                a53 = h * (64448.f/6561.f), a54 = h * (-212.f/729.f);
    const float a61 = h * (9017.f/3168.f),  a62 = h * (-355.f/33.f),
                a63 = h * (46732.f/5247.f), a64 = h * (49.f/176.f),
                a65 = h * (-5103.f/18656.f);
    const float ub1 = h * (35.f/384.f),    ub3 = h * (500.f/1113.f),
                ub4 = h * (125.f/192.f),   ub5 = h * (-2187.f/6784.f),
                ub6 = h * (11.f/84.f);

    for (int step = 0; step < 8; ++step) {
        float* sdst = (step == 7) ? out : s;

        ode_stage_kernel<0,0><<<nblk, 256, stage_smem, stream>>>(
            s, nullptr, nullptr, nullptr, nullptr, nullptr,
            0.f, 0.f, 0.f, 0.f, 0.f,
            Wt1, b1, Wt2, b2, Wt3, b3, k1, nullptr,
            0.f, 0.f, 0.f, 0.f, 0.f);
        ode_stage_kernel<1,0><<<nblk, 256, stage_smem, stream>>>(
            s, k1, nullptr, nullptr, nullptr, nullptr,
            a21, 0.f, 0.f, 0.f, 0.f,
            Wt1, b1, Wt2, b2, Wt3, b3, k2, nullptr,
            0.f, 0.f, 0.f, 0.f, 0.f);
        ode_stage_kernel<2,0><<<nblk, 256, stage_smem, stream>>>(
            s, k1, k2, nullptr, nullptr, nullptr,
            a31, a32, 0.f, 0.f, 0.f,
            Wt1, b1, Wt2, b2, Wt3, b3, k3, nullptr,
            0.f, 0.f, 0.f, 0.f, 0.f);
        ode_stage_kernel<3,0><<<nblk, 256, stage_smem, stream>>>(
            s, k1, k2, k3, nullptr, nullptr,
            a41, a42, a43, 0.f, 0.f,
            Wt1, b1, Wt2, b2, Wt3, b3, k4, nullptr,
            0.f, 0.f, 0.f, 0.f, 0.f);
        ode_stage_kernel<4,0><<<nblk, 256, stage_smem, stream>>>(
            s, k1, k2, k3, k4, nullptr,
            a51, a52, a53, a54, 0.f,
            Wt1, b1, Wt2, b2, Wt3, b3, k5, nullptr,
            0.f, 0.f, 0.f, 0.f, 0.f);
        ode_stage_kernel<5,1><<<nblk, 256, stage_smem, stream>>>(
            s, k1, k2, k3, k4, k5,
            a61, a62, a63, a64, a65,
            Wt1, b1, Wt2, b2, Wt3, b3, nullptr, sdst,
            ub1, ub3, ub4, ub5, ub6);
    }
}
